// Loss_72980084294067
// MI455X (gfx1250) — matrix operand order repaired and checked by reading, not compiled, not run on hardware
//
#include <hip/hip_runtime.h>
#include <hip/hip_bf16.h>

typedef __attribute__((ext_vector_type(16))) _Float16 v16h;
typedef __attribute__((ext_vector_type(8)))  _Float16 v8h;
typedef __attribute__((ext_vector_type(8)))  float    v8f;
typedef __attribute__((ext_vector_type(4)))  unsigned int u32x4;
typedef __attribute__((ext_vector_type(8)))  int      i32x8;
typedef __attribute__((ext_vector_type(4)))  int      i32x4;

#define NIMG 64
#define NREG 49
#define NTXT 64
#define NWRD 64
#define DIM  256

#define LAMBDA 10.0f
#define SLOPE  0.1f

// ---- LDS layout (units: halfs for LD_*, bytes for OFF_*) ----
#define LD_KN  264   // 256 + 8 pad halfs, row stride 528B (16B aligned)
#define LD_VT  72    // 64 + 8 pad, 144B
#define LD_AT  72
#define LD_RAW 68    // floats, 272B
#define OFF_KN   0
#define OFF_QN   34816
#define OFF_VT   0        // reuses KN/QN region after GEMM1
#define OFF_RAW  69632
#define OFF_AT   87040
#define OFF_RSCL 96256    // 64 f32
#define OFF_PSUM 96512    // 128 f32
#define OFF_SDOT 97024    // 64 f32
#define OFF_SNRM 97280    // 64 f32
#define SMEM_BYTES 98304

#if defined(__HIP_DEVICE_COMPILE__) && __has_builtin(__builtin_amdgcn_tensor_load_to_lds)
#define HAVE_TDM 1
#else
#define HAVE_TDM 0
#endif

#if HAVE_TDM
// TDM 2D tile load: tile = 64 rows x 256 halfs, LDS row stride 528B via
// pad (interval 128 DWORDs = 1 row, amount 4 DWORDs = 16B). Rows beyond
// tensor_rows read OOB -> zero-filled in LDS (used to pad QN to 64 rows).
__device__ __forceinline__ void tdm_load_tile64(unsigned lds_addr, const void* gsrc,
                                                unsigned tensor_rows) {
  unsigned long long ga = (unsigned long long)(uintptr_t)gsrc;
  u32x4 g0;
  g0[0] = 1u;                                              // count=1, user D#
  g0[1] = lds_addr;                                        // D#.lds_addr
  g0[2] = (unsigned)(ga & 0xffffffffu);                    // global_addr[31:0]
  g0[3] = (unsigned)((ga >> 32) & 0x01ffffffu)             // global_addr[56:32]
          | 0x80000000u;                                   // type=2 ("image")
  i32x8 g1;
  g1[0] = (1 << 16)        // data_size = 2 bytes
        | (1 << 20)        // pad_enable
        | (6 << 22)        // pad_interval: 128 DWORDs (one 512B row)
        | (3 << 25);       // pad_amount: 4 DWORDs (16B)
  g1[1] = (int)(256u << 16);            // tensor_dim0 = 256 elements
  g1[2] = (int)(tensor_rows << 16);     // tensor_dim1
  g1[3] = (int)(256u << 16);            // tile_dim0 = 256
  g1[4] = 64;                           // tile_dim1 = 64 rows
  g1[5] = 256;                          // tensor_dim0_stride = 256 elements
  g1[6] = 0;
  g1[7] = 0;
  i32x4 gz = {0, 0, 0, 0};
#if __clang_major__ >= 23
  i32x8 gz8 = {0, 0, 0, 0, 0, 0, 0, 0};
  __builtin_amdgcn_tensor_load_to_lds(g0, g1, gz, gz, gz8, 0);
#else
  __builtin_amdgcn_tensor_load_to_lds(g0, g1, gz, gz, 0);
#endif
}
#endif

// A-matrix 16x32 f16 fragment (ISA 7.12.2): lanes hold row M=lane&15;
// lanes 0-15 take K=[0..7]+[16..23], lanes 16-31 take K=[8..15]+[24..31].
__device__ __forceinline__ v16h frag_a16(const _Float16* row, int kbase, int lane) {
  const _Float16* p = row + kbase + ((lane & 16) ? 8 : 0);
  v8h lo = *(const v8h*)p;
  v8h hi = *(const v8h*)(p + 16);
  v16h r;
#pragma unroll
  for (int e = 0; e < 8; ++e) { r[e] = lo[e]; r[e + 8] = hi[e]; }
  return r;
}

// B-matrix 32x16 f16 fragment: lane selects column N=lane&15; lanes 0-15
// hold K=[0..7]+[16..23], lanes 16-31 hold K=[8..15]+[24..31]. `row` points at the N-major
// storage row for this column, so both chunks are contiguous.
__device__ __forceinline__ v16h frag_b16(const _Float16* row, int kbase, int lane) {
  const _Float16* p = row + kbase + ((lane & 16) ? 8 : 0);
  v8h lo = *(const v8h*)p;
  v8h hi = *(const v8h*)(p + 16);
  v16h r;
#pragma unroll
  for (int e = 0; e < 8; ++e) { r[e] = lo[e]; r[e + 8] = hi[e]; }
  return r;
}

// Pre-pass: l2-normalize query/key rows -> f16; text value -> f16;
// img value row / max(||row||,1e-8) -> f32 (vhat).
__global__ void scan_prep(const float* __restrict__ imgq, const float* __restrict__ imgv,
                          const float* __restrict__ txtk, const float* __restrict__ txtv,
                          _Float16* __restrict__ qn_h, _Float16* __restrict__ kn_h,
                          _Float16* __restrict__ val_h, float* __restrict__ vhat) {
  const int IQ = NIMG * NREG, TS = NTXT * NWRD;
  int row = blockIdx.x, tid = threadIdx.x;
  __shared__ float wsum[8];
  int cls; const float* src; size_t r;
  if (row < IQ)             { cls = 0; r = row;                src = imgq + r * DIM; }
  else if (row < IQ + TS)   { cls = 1; r = row - IQ;           src = txtk + r * DIM; }
  else if (row < IQ + 2*TS) { cls = 2; r = row - IQ - TS;      src = txtv + r * DIM; }
  else                      { cls = 3; r = row - IQ - 2 * TS;  src = imgv + r * DIM; }
  float x = src[tid];
  if (cls == 2) { val_h[r * DIM + tid] = (_Float16)x; return; }
  float ss = x * x;
#pragma unroll
  for (int m = 1; m < 32; m <<= 1) ss += __shfl_xor(ss, m, 32);
  if ((tid & 31) == 0) wsum[tid >> 5] = ss;
  __syncthreads();
  float tot = 0.f;
#pragma unroll
  for (int wv = 0; wv < 8; ++wv) tot += wsum[wv];
  float nrm = sqrtf(tot);
  if (cls == 0)      qn_h[r * DIM + tid] = (_Float16)(x / fmaxf(nrm, 1e-12f));
  else if (cls == 1) kn_h[r * DIM + tid] = (_Float16)(x / fmaxf(nrm, 1e-12f));
  else               vhat[r * DIM + tid] = x / fmaxf(nrm, 1e-8f);
}

// One workgroup per (txt t, img i) pair. 128 threads = 4 wave32.
__global__ void __launch_bounds__(128)
scan_i2t(const _Float16* __restrict__ qn_h, const _Float16* __restrict__ kn_h,
         const _Float16* __restrict__ val_h, const float* __restrict__ vhat,
         const int* __restrict__ text_len, float* __restrict__ out) {
  extern __shared__ char smem[];
  const int t = blockIdx.x, i = blockIdx.y;
  const int tid = threadIdx.x, lane = tid & 31, wv = tid >> 5;
  const int lane15 = lane & 15;
  const int mrow = (lane & 16) ? 8 : 0;   // C/D tile: lanes 16-31 hold M = r+8

  _Float16* KN = (_Float16*)(smem + OFF_KN);
  _Float16* QN = (_Float16*)(smem + OFF_QN);
  _Float16* VT = (_Float16*)(smem + OFF_VT);
  _Float16* AT = (_Float16*)(smem + OFF_AT);
  float* RAW  = (float*)(smem + OFF_RAW);
  float* RSCL = (float*)(smem + OFF_RSCL);
  float* PSUM = (float*)(smem + OFF_PSUM);
  float* SDOT = (float*)(smem + OFF_SDOT);
  float* SNRM = (float*)(smem + OFF_SNRM);

  // ---- stage KN = kn[t] (64x256 f16), QN = qn[i] (rows >= 49 zeroed) ----
  const _Float16* ksrc = kn_h + (size_t)t * NWRD * DIM;
  const _Float16* qsrc = qn_h + (size_t)i * NREG * DIM;
#if HAVE_TDM
  if (wv == 0) {
    unsigned lds0 = (unsigned)(unsigned long long)
        (__attribute__((address_space(3))) char*)smem;
    tdm_load_tile64(lds0 + OFF_KN, ksrc, NWRD);   // full 64-row tensor
    tdm_load_tile64(lds0 + OFF_QN, qsrc, NREG);   // rows 49..63 OOB -> zeros
    __builtin_amdgcn_s_wait_tensorcnt(0);
  }
#else
  for (int e = tid * 8; e < NWRD * DIM; e += 128 * 8) {
    int s = e >> 8, d = e & (DIM - 1);
    *(v8h*)(KN + s * LD_KN + d) = *(const v8h*)(ksrc + s * DIM + d);
    v8h qv;
    if (s < NREG) {
      qv = *(const v8h*)(qsrc + s * DIM + d);
    } else {
#pragma unroll
      for (int e2 = 0; e2 < 8; ++e2) qv[e2] = (_Float16)0.f;
    }
    *(v8h*)(QN + s * LD_KN + d) = qv;
  }
#endif
  __syncthreads();

  // ---- GEMM1: raw[s][q] = kn[t] . qn[i]^T  (64x64, K=256) ----
  {
    const int sbase = wv * 16;
    v8f acc[4] = {};
#pragma unroll
    for (int kk = 0; kk < DIM; kk += 32) {
      v16h a = frag_a16(KN + (sbase + lane15) * LD_KN, kk, lane);
#pragma unroll
      for (int qt = 0; qt < 4; ++qt) {
        v16h b = frag_b16(QN + (qt * 16 + lane15) * LD_KN, kk, lane);
        acc[qt] = __builtin_amdgcn_wmma_f32_16x16x32_f16(false, a, false, b,
                                                         (short)0, acc[qt], false, false);
      }
    }
#pragma unroll
    for (int qt = 0; qt < 4; ++qt)
#pragma unroll
      for (int rr = 0; rr < 8; ++rr)
        RAW[(sbase + rr + mrow) * LD_RAW + qt * 16 + lane15] = acc[qt][rr];
  }

  // warm the epilogue's vhat block (global_prefetch_b8) while attn post-proc runs
  {
    const float* vbase = vhat + (size_t)i * NREG * DIM;
    for (int off = tid * 32; off < NREG * DIM; off += 128 * 32)
      __builtin_prefetch(vbase + off, 0, 1);
  }
  __syncthreads();

  // ---- LeakyReLU(0.1) + per-s l2norm over q (padded q cols are exactly 0) ----
  {
    int s = tid & 63, half = tid >> 6;
    float ss = 0.f;
    for (int q = half * 32; q < half * 32 + 32; ++q) {
      float v = RAW[s * LD_RAW + q];
      v = v > 0.f ? v : SLOPE * v;
      RAW[s * LD_RAW + q] = v;
      ss += v * v;
    }
    PSUM[half * 64 + s] = ss;
  }
  __syncthreads();
  if (tid < 64) RSCL[tid] = 1.f / fmaxf(sqrtf(PSUM[tid] + PSUM[64 + tid]), 1e-12f);
  __syncthreads();

  // ---- stage VT = txt value, transposed to d-major (overwrites KN/QN region) ----
  const _Float16* vsrc = val_h + (size_t)t * NWRD * DIM;
  for (int e = tid; e < NWRD * DIM; e += 128) {
    int s = e >> 8, d = e & (DIM - 1);
    VT[d * LD_VT + s] = vsrc[s * DIM + d];
  }

  // ---- masked softmax (lambda=10) over s + focal_equal; AT[q][s] f16 ----
  int len = text_len[t];
  float nw = (float)len;
  if (tid < 64) {
    int q = tid;
    if (q < NREG) {
      float m = -1e30f;
      for (int s = 0; s < len; ++s) m = fmaxf(m, RAW[s * LD_RAW + q] * RSCL[s]);
      float Z = 0.f;
      for (int s = 0; s < len; ++s) Z += __expf(LAMBDA * (RAW[s * LD_RAW + q] * RSCL[s] - m));
      float invZ = 1.f / Z;
      float Z2 = 0.f;
      for (int s = 0; s < len; ++s) {
        float a = __expf(LAMBDA * (RAW[s * LD_RAW + q] * RSCL[s] - m)) * invZ;
        if (a * nw - 1.f > 0.f) Z2 += a;   // sum(attn)==1 -> funcF = a*n_word - 1
      }
      float invZ2 = 1.f / Z2;
      for (int s = 0; s < NWRD; ++s) {
        float o = 0.f;
        if (s < len) {
          float a = __expf(LAMBDA * (RAW[s * LD_RAW + q] * RSCL[s] - m)) * invZ;
          if (a * nw - 1.f > 0.f) o = a * invZ2;
        }
        AT[q * LD_AT + s] = (_Float16)o;
      }
    } else {
      for (int s = 0; s < NWRD; ++s) AT[q * LD_AT + s] = (_Float16)0.f;
    }
  }
  __syncthreads();

  // ---- GEMM2: wei[q][d] = AT . val[t]  (64x256, K=64), wei kept in regs,
  //      fused cosine reduction vs vhat ----
  {
    const int qbase = wv * 16;
    v8f acc[16] = {};
#pragma unroll
    for (int kk = 0; kk < NWRD; kk += 32) {
      v16h a = frag_a16(AT + (qbase + lane15) * LD_AT, kk, lane);
#pragma unroll
      for (int nt = 0; nt < 16; ++nt) {
        v16h b = frag_b16(VT + (nt * 16 + lane15) * LD_VT, kk, lane);
        acc[nt] = __builtin_amdgcn_wmma_f32_16x16x32_f16(false, a, false, b,
                                                         (short)0, acc[nt], false, false);
      }
    }
    float pd[8], pn[8];
#pragma unroll
    for (int rr = 0; rr < 8; ++rr) { pd[rr] = 0.f; pn[rr] = 0.f; }
#pragma unroll
    for (int nt = 0; nt < 16; ++nt) {
      int d = nt * 16 + lane15;
#pragma unroll
      for (int rr = 0; rr < 8; ++rr) {
        int q = qbase + rr + mrow;
        int qc = q < NREG ? q : NREG - 1;          // padded q rows ignored later
        float w = acc[nt][rr];
        float vh = vhat[((size_t)i * NREG + qc) * DIM + d];
        pd[rr] += vh * w;
        pn[rr] += w * w;
      }
    }
#pragma unroll
    for (int rr = 0; rr < 8; ++rr) {
      float d0 = pd[rr], n0 = pn[rr];
#pragma unroll
      for (int m = 1; m < 16; m <<= 1) {
        d0 += __shfl_xor(d0, m, 32);
        n0 += __shfl_xor(n0, m, 32);
      }
      if (lane15 == 0) { int q = qbase + rr + mrow; SDOT[q] = d0; SNRM[q] = n0; }
    }
  }
  __syncthreads();

  if (tid == 0) {
    float a = 0.f;
    for (int q = 0; q < NREG; ++q)
      a += SDOT[q] / fmaxf(sqrtf(SNRM[q]), 1e-12f);
    out[(size_t)i * NTXT + t] = a / (float)NREG;   // i2t.T -> [n_img, n_txt]
  }
}

extern "C" void kernel_launch(void* const* d_in, const int* in_sizes, int n_in,
                              void* d_out, int out_size, void* d_ws, size_t ws_size,
                              hipStream_t stream) {
  const float* imgq = (const float*)d_in[0];
  const float* imgv = (const float*)d_in[1];
  const float* txtk = (const float*)d_in[2];
  const float* txtv = (const float*)d_in[3];
  const int*   tlen = (const int*)d_in[4];
  float* out = (float*)d_out;

  char* ws = (char*)d_ws;
  const size_t IQ = (size_t)NIMG * NREG, TS = (size_t)NTXT * NWRD;
  _Float16* qn_h  = (_Float16*)ws;                                  // IQ*D halfs
  _Float16* kn_h  = (_Float16*)(ws + IQ * DIM * 2);                 // TS*D halfs
  _Float16* val_h = (_Float16*)(ws + (IQ + TS) * DIM * 2);          // TS*D halfs
  float*    vhat  = (float*)   (ws + (IQ + 2 * TS) * DIM * 2);      // IQ*D f32
  (void)ws_size; (void)in_sizes; (void)n_in; (void)out_size;

  int rows = (int)(IQ + 2 * TS + IQ);
  scan_prep<<<rows, 256, 0, stream>>>(imgq, imgv, txtk, txtv, qn_h, kn_h, val_h, vhat);

  dim3 grid(NTXT, NIMG);
  scan_i2t<<<grid, 128, SMEM_BYTES, stream>>>(qn_h, kn_h, val_h, vhat, tlen, out);
}
